// VoxelBackBone8x_v2_79620103733510
// MI455X (gfx1250) — compile-verified, matrix operand order checked by reading, not run on hardware
//
#include <hip/hip_runtime.h>
#include <hip/hip_bf16.h>

// ---------- types ----------
typedef __attribute__((ext_vector_type(16))) __bf16        v16bf;
typedef __attribute__((ext_vector_type(8)))  float         v8f;
typedef __attribute__((ext_vector_type(8)))  unsigned int  v8u;
typedef __attribute__((ext_vector_type(4)))  unsigned int  v4u;
typedef __attribute__((ext_vector_type(2)))  unsigned int  v2u;

__device__ __forceinline__ unsigned short f2bf(float f) {
    unsigned int u = __float_as_uint(f);
    u += 0x7FFFu + ((u >> 16) & 1u);   // round-to-nearest-even
    return (unsigned short)(u >> 16);
}

// ---------- utility kernels ----------
__global__ void zero_u32(unsigned int* __restrict__ p, long long n) {
    long long i = (long long)blockIdx.x * blockDim.x + threadIdx.x;
    long long st = (long long)gridDim.x * blockDim.x;
    for (; i < n; i += st) p[i] = 0u;
}

__global__ void scatter_voxels(const float* __restrict__ f, const int* __restrict__ c,
                               unsigned short* __restrict__ x0, unsigned char* __restrict__ m0,
                               int N) {
    int i = blockIdx.x * blockDim.x + threadIdx.x;
    if (i >= N) return;
    int z = c[i * 4 + 1], y = c[i * 4 + 2], x = c[i * 4 + 3];
    size_t p = ((size_t)z * 256 + y) * 256 + x;
    for (int ch = 0; ch < 4; ++ch) x0[p * 4 + ch] = f2bf(f[i * 4 + ch]);
    m0[p] = 1;
}

// grid = ((Wo+255)/256, Ho, Do); no integer division anywhere.
__global__ void dilate_mask(const unsigned char* __restrict__ mi, unsigned char* __restrict__ mo,
                            int Di, int Hi, int Wi, int Ho, int Wo,
                            int kd, int kh, int kw, int sz, int sy, int sx,
                            int pz, int py, int px) {
    int x = blockIdx.x * 256 + threadIdx.x;
    if (x >= Wo) return;
    int y = blockIdx.y;
    int z = blockIdx.z;
    int on = 0;
    for (int a = 0; a < kd; ++a) {
        int zi = z * sz + a - pz; if (zi < 0 || zi >= Di) continue;
        for (int b = 0; b < kh; ++b) {
            int yi = y * sy + b - py; if (yi < 0 || yi >= Hi) continue;
            for (int cc = 0; cc < kw; ++cc) {
                int xi = x * sx + cc - px; if (xi < 0 || xi >= Wi) continue;
                on |= mi[((size_t)zi * Hi + yi) * Wi + xi];
            }
        }
    }
    mo[((size_t)z * Ho + y) * Wo + x] = on ? 1 : 0;
}

// Pack weights (tap,cin,cout) f32 -> per-tile B operand layout, bf16.
// Tile = (tap, kchunk, ntile): 32 lanes x 16 bf16.
// lane = 16*hi + n : column N = n, K locals {hi*8+j | j<8} then {16+hi*8+(j-8)}.
// NT and KC are powers of two; shifts/masks passed from host.
__global__ void prepack_w(const float* __restrict__ W, unsigned short* __restrict__ out,
                          long long total, int cin, int cout,
                          int ntMask, int ntShift, int kcMask, int kcShift) {
    long long i = (long long)blockIdx.x * blockDim.x + threadIdx.x;
    if (i >= total) return;
    int j    = (int)(i & 15);
    int lane = (int)((i >> 4) & 31);
    int tile = (int)(i >> 9);
    int nt  = tile & ntMask;
    int t2  = tile >> ntShift;
    int kc  = t2 & kcMask;
    int tap = t2 >> kcShift;
    int hi = lane >> 4, n = lane & 15;
    int kl = (j < 8) ? (hi * 8 + j) : (16 + hi * 8 + (j - 8));
    int c  = kc * 32 + kl;
    int ng = nt * 16 + n;
    float v = (c < cin) ? W[((size_t)tap * cin + c) * cout + ng] : 0.0f;
    out[i] = f2bf(v);
}

// ---------- implicit-GEMM conv + BN + ReLU + mask, bf16 WMMA ----------
// grid = (XT, Ho, Do*NT), block = 32 (one wave).
template <int CI, int CO, int KD, int KH, int KW, bool LAST>
__global__ __launch_bounds__(32)
void conv_wmma(const unsigned short* __restrict__ xin,
               const unsigned short* __restrict__ wpk,
               const float* __restrict__ gamma, const float* __restrict__ beta,
               const unsigned char* __restrict__ mask,
               void* __restrict__ outp,
               int Di, int Hi, int Wi,
               int Ho, int Wo,
               int sz, int sy, int sx, int pz, int py, int px) {
    constexpr int KC  = (CI + 31) / 32;
    constexpr int NT  = CO / 16;
    constexpr int NTS = (NT == 1) ? 0 : (NT == 2) ? 1 : (NT == 4) ? 2 : 3;
    constexpr int REM = (CI >= 32) ? 32 : CI;     // K valid per 32-chunk (4, 16 or 32)

    const int lane = threadIdx.x;
    const int hi = lane >> 4, lo = lane & 15;
    const int xt = blockIdx.x;
    const int y  = blockIdx.y;
    const int nt = (int)(blockIdx.z & (NT - 1));
    const int z  = (int)(blockIdx.z >> NTS);
    const int xo = xt * 16 + lo;                  // A-matrix row for this lane pair
    const int g0 = hi * 8;

    v8f acc = {};
    for (int kz = 0; kz < KD; ++kz) {
        const int zi = z * sz + kz - pz;
        const bool zOK = (zi >= 0) && (zi < Di);
        for (int ky = 0; ky < KH; ++ky) {
            const int yi = y * sy + ky - py;
            if (!zOK || yi < 0 || yi >= Hi) continue;     // uniform skip
            const size_t planeOff = ((size_t)zi * Hi + yi) * (size_t)Wi;
#pragma unroll
            for (int kx = 0; kx < KW; ++kx) {
                const int tap = (kz * KH + ky) * KW + kx;
                const int xi  = xo * sx + kx - px;
                const bool rowOK = (xi >= 0) && (xi < Wi);
#pragma unroll
                for (int kc = 0; kc < KC; ++kc) {
                    v8u au = {};
                    if (rowOK) {
                        const unsigned short* cp = xin + (planeOff + xi) * CI + kc * 32;
                        if constexpr (REM == 32) {
                            v4u t0 = *(const v4u*)(cp + g0);
                            v4u t1 = *(const v4u*)(cp + 16 + g0);
                            au[0] = t0[0]; au[1] = t0[1]; au[2] = t0[2]; au[3] = t0[3];
                            au[4] = t1[0]; au[5] = t1[1]; au[6] = t1[2]; au[7] = t1[3];
                        } else if constexpr (REM == 16) {
                            v4u t0 = *(const v4u*)(cp + g0);   // hi=0: K0..7, hi=1: K8..15
                            au[0] = t0[0]; au[1] = t0[1]; au[2] = t0[2]; au[3] = t0[3];
                        } else {                               // REM == 4
                            if (hi == 0) {
                                v2u t0 = *(const v2u*)cp;      // K0..3
                                au[0] = t0[0]; au[1] = t0[1];
                            }
                        }
                    }
                    const unsigned short* bp =
                        wpk + ((size_t)((tap * KC + kc) * NT + nt) * 32 + lane) * 16;
                    v4u b0 = *(const v4u*)bp;
                    v4u b1 = *(const v4u*)(bp + 8);
                    v8u bu;
                    bu[0] = b0[0]; bu[1] = b0[1]; bu[2] = b0[2]; bu[3] = b0[3];
                    bu[4] = b1[0]; bu[5] = b1[1]; bu[6] = b1[2]; bu[7] = b1[3];

                    v16bf a = __builtin_bit_cast(v16bf, au);
                    v16bf b = __builtin_bit_cast(v16bf, bu);
                    // EXEC all-1s at this point (divergent loads reconverged).
                    acc = __builtin_amdgcn_wmma_f32_16x16x32_bf16(
                        false, a, false, b, (short)0, acc, false, false);
                }
            }
        }
    }

    // epilogue: y = relu(acc * gamma/sqrt(1+eps) + beta) * mask
    const int nc = nt * 16 + lo;
    const float g  = gamma[nc] * 0.99950037f;     // 1/sqrt(1.001)
    const float bb = beta[nc];
    const size_t rowBase = ((size_t)z * Ho + y) * (size_t)Wo;
    const unsigned char* mrow = mask + rowBase;
#pragma unroll
    for (int r = 0; r < 8; ++r) {
        const int x = xt * 16 + r + hi * 8;
        float v = acc[r] * g + bb;
        v = v > 0.0f ? v : 0.0f;
        v *= (float)mrow[x];
        const size_t o = (rowBase + x) * (size_t)CO + nc;
        if constexpr (LAST) ((float*)outp)[o] = v;
        else                ((unsigned short*)outp)[o] = f2bf(v);
    }
}

// ---------- host orchestration ----------
struct Spec { int kd, kh, kw, cin, cout, sz, sy, sx, pz, py, px, subm; };
static const Spec g_specs[12] = {
    {3,3,3,  4, 16, 1,1,1, 1,1,1, 1},
    {3,3,3, 16, 16, 1,1,1, 1,1,1, 0},
    {3,3,3, 16, 32, 2,2,2, 1,1,1, 0},
    {3,3,3, 32, 32, 1,1,1, 1,1,1, 1},
    {3,3,3, 32, 32, 1,1,1, 1,1,1, 0},
    {3,3,3, 32, 64, 2,2,2, 1,1,1, 0},
    {3,3,3, 64, 64, 1,1,1, 1,1,1, 1},
    {3,3,3, 64, 64, 1,1,1, 1,1,1, 1},
    {3,3,3, 64, 64, 2,2,2, 0,1,1, 0},
    {3,3,3, 64, 64, 1,1,1, 1,1,1, 1},
    {3,3,3, 64, 64, 1,1,1, 1,1,1, 1},
    {3,1,1, 64,128, 2,1,1, 0,0,0, 0},
};

static int ilog2i(int v) { int s = 0; while ((1 << s) < v) ++s; return s; }

template <int CI, int CO, int KD, int KH, int KW, bool LAST>
static void launch_conv(const unsigned short* xin, const unsigned short* wpk,
                        const float* g, const float* b, const unsigned char* m,
                        void* out, int Di, int Hi, int Wi, int Do, int Ho, int Wo,
                        const Spec& s, hipStream_t st) {
    constexpr int NT = CO / 16;
    dim3 grid((unsigned)(Wo / 16), (unsigned)Ho, (unsigned)(Do * NT));
    conv_wmma<CI, CO, KD, KH, KW, LAST><<<grid, 32, 0, st>>>(
        xin, wpk, g, b, m, out, Di, Hi, Wi, Ho, Wo,
        s.sz, s.sy, s.sx, s.pz, s.py, s.px);
}

extern "C" void kernel_launch(void* const* d_in, const int* in_sizes, int n_in,
                              void* d_out, int out_size, void* d_ws, size_t ws_size,
                              hipStream_t stream) {
    (void)in_sizes; (void)n_in; (void)out_size; (void)ws_size;
    const float* feats  = (const float*)d_in[0];
    const float* Ws[12];  for (int l = 0; l < 12; ++l) Ws[l]  = (const float*)d_in[1 + l];
    const float* gms[12]; for (int l = 0; l < 12; ++l) gms[l] = (const float*)d_in[13 + l];
    const float* bts[12]; for (int l = 0; l < 12; ++l) bts[l] = (const float*)d_in[25 + l];
    const int*   coords = (const int*)d_in[37];

    char* ws = (char*)d_ws;
    unsigned short* actA  = (unsigned short*)ws;                          // 64 MB
    unsigned short* actB  = (unsigned short*)(ws + ((size_t)64 << 20));   // 64 MB
    unsigned char*  maskA = (unsigned char*)(ws + ((size_t)128 << 20));   // 2 MB
    unsigned char*  maskB = (unsigned char*)(ws + ((size_t)130 << 20));   // 2 MB
    unsigned short* wpk   = (unsigned short*)(ws + ((size_t)132 << 20));

    // per-layer packed-weight offsets (in ushorts)
    size_t woff[12], wacc = 0;
    for (int l = 0; l < 12; ++l) {
        const Spec& s = g_specs[l];
        int taps = s.kd * s.kh * s.kw;
        int KC = (s.cin + 31) / 32, NT = s.cout / 16;
        woff[l] = wacc;
        wacc += (size_t)taps * KC * NT * 32 * 16;
    }

    // zero dense input grid + mask (ws poisoned by harness; re-init every call)
    zero_u32<<<2048, 256, 0, stream>>>((unsigned int*)actA,
                                       (long long)25 * 256 * 256 * 4 * 2 / 4);
    zero_u32<<<1024, 256, 0, stream>>>((unsigned int*)maskA,
                                       (long long)25 * 256 * 256 / 4);
    scatter_voxels<<<(40000 + 255) / 256, 256, 0, stream>>>(feats, coords, actA, maskA, 40000);

    // prepack all layer weights
    for (int l = 0; l < 12; ++l) {
        const Spec& s = g_specs[l];
        int taps = s.kd * s.kh * s.kw;
        int KC = (s.cin + 31) / 32, NT = s.cout / 16;
        long long tot = (long long)taps * KC * NT * 32 * 16;
        prepack_w<<<(unsigned)((tot + 255) / 256), 256, 0, stream>>>(
            Ws[l], wpk + woff[l], tot, s.cin, s.cout,
            NT - 1, ilog2i(NT), KC - 1, ilog2i(KC));
    }

    // layer pipeline
    int Di = 25, Hi = 256, Wi = 256;
    unsigned short* cur = actA; unsigned short* nxt = actB;
    unsigned char*  mcur = maskA; unsigned char* mnxt = maskB;
    for (int l = 0; l < 12; ++l) {
        const Spec& s = g_specs[l];
        int Do = (Di + 2 * s.pz - s.kd) / s.sz + 1;
        int Ho = (Hi + 2 * s.py - s.kh) / s.sy + 1;
        int Wo = (Wi + 2 * s.px - s.kw) / s.sx + 1;
        if (!s.subm) {
            dim3 dg((unsigned)((Wo + 255) / 256), (unsigned)Ho, (unsigned)Do);
            dilate_mask<<<dg, 256, 0, stream>>>(
                mcur, mnxt, Di, Hi, Wi, Ho, Wo,
                s.kd, s.kh, s.kw, s.sz, s.sy, s.sx, s.pz, s.py, s.px);
            unsigned char* tm = mcur; mcur = mnxt; mnxt = tm;
        }
        const unsigned short* wl = wpk + woff[l];
        void* out = (l == 11) ? d_out : (void*)nxt;
        switch (l) {
        case 0:  launch_conv< 4, 16,3,3,3,false>(cur, wl, gms[l], bts[l], mcur, out, Di,Hi,Wi,Do,Ho,Wo, s, stream); break;
        case 1:  launch_conv<16, 16,3,3,3,false>(cur, wl, gms[l], bts[l], mcur, out, Di,Hi,Wi,Do,Ho,Wo, s, stream); break;
        case 2:  launch_conv<16, 32,3,3,3,false>(cur, wl, gms[l], bts[l], mcur, out, Di,Hi,Wi,Do,Ho,Wo, s, stream); break;
        case 3:
        case 4:  launch_conv<32, 32,3,3,3,false>(cur, wl, gms[l], bts[l], mcur, out, Di,Hi,Wi,Do,Ho,Wo, s, stream); break;
        case 5:  launch_conv<32, 64,3,3,3,false>(cur, wl, gms[l], bts[l], mcur, out, Di,Hi,Wi,Do,Ho,Wo, s, stream); break;
        case 6: case 7: case 8: case 9: case 10:
                 launch_conv<64, 64,3,3,3,false>(cur, wl, gms[l], bts[l], mcur, out, Di,Hi,Wi,Do,Ho,Wo, s, stream); break;
        case 11: launch_conv<64,128,3,1,1,true >(cur, wl, gms[l], bts[l], mcur, out, Di,Hi,Wi,Do,Ho,Wo, s, stream); break;
        }
        unsigned short* ta = cur; cur = nxt; nxt = ta;
        Di = Do; Hi = Ho; Wi = Wo;
    }
}
